// MCB_78374563217714
// MI455X (gfx1250) — matrix operand order repaired and checked by reading, not compiled, not run on hardware
//
#include <hip/hip_runtime.h>
#include <hip/hip_bf16.h>

#define B_SZ  512
#define D_IN  2048
#define MCB   16000
#define OUT_N 3000

typedef __attribute__((ext_vector_type(16))) __bf16 v16bf;
typedef __attribute__((ext_vector_type(8)))  __bf16 v8bf;
typedef __attribute__((ext_vector_type(8)))  float  v8f;

// ---------------------------------------------------------------------------
// Kernel 1: fused count-sketch(x1) + circular convolution (no FFT).
//   z[b,k] = sum_i s1[i]*x0[b,i] * p2[b, (k - h1[i]) mod MCB]
// p2[b,:] (64000 B) lives entirely in LDS. Scatter is made deterministic by
// giving each thread ownership of bins with (h & 255) == tid (single writer
// per bin, fixed j order -> bitwise-reproducible across graph replays).
// Grid: (4 k-slices, 512 batch rows), 256 threads. Output z stored as bf16.
// ---------------------------------------------------------------------------
__global__ __launch_bounds__(256) void mcb_conv_kernel(
    const float* __restrict__ x0, const float* __restrict__ x1,
    const float* __restrict__ s1, const float* __restrict__ s2,
    const int* __restrict__ h1,  const int* __restrict__ h2,
    __bf16* __restrict__ z)
{
    __shared__ float p2s[MCB];              // 64000 bytes
    const int tid   = threadIdx.x;
    const int b     = blockIdx.y;
    const int kbase = blockIdx.x * (MCB / 4);
    const int kend  = kbase + (MCB / 4);

    for (int k = tid; k < MCB; k += 256) p2s[k] = 0.0f;
    __syncthreads();

    // deterministic count-sketch scatter of x1 row b into LDS
    for (int j = 0; j < D_IN; ++j) {
        int h = h2[j];                      // uniform -> scalar loads
        if ((h & 255) == tid) {
            p2s[h] += s2[j] * x1[(size_t)b * D_IN + j];
        }
    }
    __syncthreads();

    const int KT = 16;                      // 16 k-values per thread
    float acc[KT];
    #pragma unroll
    for (int t = 0; t < KT; ++t) acc[t] = 0.0f;

    const int kb = kbase + tid;
    for (int i = 0; i < D_IN; ++i) {
        const int   hi = h1[i];                                 // uniform
        const float ai = s1[i] * x0[(size_t)b * D_IN + i];      // uniform
        int off0 = kb - hi;
        if (off0 < 0) off0 += MCB;          // off0 in [0, MCB)
        #pragma unroll
        for (int t = 0; t < KT; ++t) {
            int off = off0 + t * 256;       // < 2*MCB -> single wrap
            if (off >= MCB) off -= MCB;
            acc[t] += ai * p2s[off];
        }
    }

    #pragma unroll
    for (int t = 0; t < KT; ++t) {
        int k = kb + t * 256;
        if (k < kend) z[(size_t)b * MCB + k] = (__bf16)acc[t];
    }
}

// ---------------------------------------------------------------------------
// Kernel 2: out = relu(z @ W.T + bias) with bf16 WMMA, fp32 accumulate.
// M=512, N=3000 (padded to 3072 tiles), K=16000. Block tile 128x128x32,
// 8 waves (wave32) arranged 2x4, each wave owns 4x2 fragments of 16x16.
// LDS: As[128][32] bf16 (z tile, staged with GLOBAL_LOAD_ASYNC_TO_LDS_B128),
//      Bs[128][32] bf16 (W tile, fp32->bf16 converted in flight).
// ---------------------------------------------------------------------------
__global__ __launch_bounds__(256) void mcb_gemm_kernel(
    const __bf16* __restrict__ z,     // [512, 16000] bf16
    const float*  __restrict__ W,     // [3000, 16000] fp32
    const float*  __restrict__ bias,  // [3000]
    float*        __restrict__ out)   // [512, 3000]
{
    __shared__ __bf16 As[128 * 32];   // 8 KB
    __shared__ __bf16 Bs[128 * 32];   // 8 KB

    const int tid  = threadIdx.x;
    const int lane = tid & 31;        // wave32
    const int wave = tid >> 5;        // 0..7
    const int wm   = wave >> 2;       // 0..1  (M direction)
    const int wn   = wave & 3;        // 0..3  (N direction)
    const int m0   = blockIdx.y * 128;
    const int n0   = blockIdx.x * 128;

    const int halfsel = (lane < 16) ? 0 : 1;
    const int l16     = lane & 15;

    v8f acc[4][2];
    const v8f vzero = {0.f, 0.f, 0.f, 0.f, 0.f, 0.f, 0.f, 0.f};
    #pragma unroll
    for (int a = 0; a < 4; ++a)
        #pragma unroll
        for (int c = 0; c < 2; ++c) acc[a][c] = vzero;

    for (int kc = 0; kc < MCB; kc += 32) {
        // ---- stage A tile: 128x32 bf16, 512 x 16B chunks via async-to-LDS
        // (ASYNCcnt-tracked DMA path: no VGPR round trip)
        #pragma unroll
        for (int q = 0; q < 2; ++q) {
            int e   = tid + q * 256;        // 512 chunks of 8 bf16
            int row = e >> 2;
            int col = (e & 3) << 3;
            unsigned long long ga =
                (unsigned long long)(uintptr_t)(z + (size_t)(m0 + row) * MCB + kc + col);
            // generic "shared" pointer: addr[31:0] is the LDS byte offset
            unsigned lp = (unsigned)(uintptr_t)(&As[row * 32 + col]);
            asm volatile("global_load_async_to_lds_b128 %0, %1, off"
                         :: "v"(lp), "v"(ga) : "memory");
        }
        // ---- stage B tile: 128x32, fp32 -> bf16 convert, N-edge zero-fill
        #pragma unroll
        for (int q = 0; q < 4; ++q) {
            int e   = tid + q * 256;        // 1024 chunks of 4 floats
            int row = e >> 3;
            int col = (e & 7) << 2;
            int n   = n0 + row;
            float4 v = make_float4(0.f, 0.f, 0.f, 0.f);
            if (n < OUT_N)
                v = *(const float4*)(W + (size_t)n * MCB + kc + col);
            __bf16* d = &Bs[row * 32 + col];
            d[0] = (__bf16)v.x; d[1] = (__bf16)v.y;
            d[2] = (__bf16)v.z; d[3] = (__bf16)v.w;
        }
        // ---- prefetch next K chunk into cache hierarchy
        if (kc + 32 < MCB) {
            __builtin_prefetch(z + (size_t)(m0 + (tid >> 1)) * MCB + kc + 32, 0, 1);
            int pn = n0 + (tid >> 1);
            if (pn < OUT_N)
                __builtin_prefetch(W + (size_t)pn * MCB + kc + 32, 0, 1);
        }
        // wait for this wave's async LDS writes, then block-wide barrier
        asm volatile("s_wait_asynccnt 0" ::: "memory");
        __syncthreads();

        // ---- build fragments per ISA 16-bit layouts
        v16bf afr[4], bfr[2];
        #pragma unroll
        for (int im = 0; im < 4; ++im) {
            // A 16x32: lane<16 holds K{0..7,16..23}; lane>=16 holds K{8..15,24..31}
            int row  = wm * 64 + im * 16 + l16;
            int koff = halfsel * 8;
            v8bf lo = *(const v8bf*)(&As[row * 32 + koff]);
            v8bf hi = *(const v8bf*)(&As[row * 32 + koff + 16]);
            v16bf a;
            #pragma unroll
            for (int e2 = 0; e2 < 8; ++e2) { a[e2] = lo[e2]; a[e2 + 8] = hi[e2]; }
            afr[im] = a;
        }
        #pragma unroll
        for (int in_ = 0; in_ < 2; ++in_) {
            // B 32x16: lane<16 holds K{0..7,16..23} of column l16; lane>=16 holds K{8..15,24..31}
            int col  = wn * 32 + in_ * 16 + l16;
            int koff = halfsel * 8;
            v8bf lo = *(const v8bf*)(&Bs[col * 32 + koff]);
            v8bf hi = *(const v8bf*)(&Bs[col * 32 + koff + 16]);
            v16bf bb;
            #pragma unroll
            for (int e2 = 0; e2 < 8; ++e2) { bb[e2] = lo[e2]; bb[e2 + 8] = hi[e2]; }
            bfr[in_] = bb;
        }

        // ---- 8 WMMAs: D = A x B + C, fp32 accumulate
        #pragma unroll
        for (int im = 0; im < 4; ++im)
            #pragma unroll
            for (int in_ = 0; in_ < 2; ++in_)
                acc[im][in_] = __builtin_amdgcn_wmma_f32_16x16x32_bf16(
                    false, afr[im], false, bfr[in_],
                    (short)0, acc[im][in_], false, false);
        __syncthreads();
    }

    // ---- epilogue: bias + ReLU, C layout: VGPR r -> M=r (lane<16) / 8+r
    #pragma unroll
    for (int in_ = 0; in_ < 2; ++in_) {
        int col = n0 + wn * 32 + in_ * 16 + l16;
        if (col >= OUT_N) continue;
        float bv = bias[col];
        #pragma unroll
        for (int im = 0; im < 4; ++im) {
            int mbase = m0 + wm * 64 + im * 16 + halfsel * 8;
            v8f a = acc[im][in_];
            #pragma unroll
            for (int r = 0; r < 8; ++r) {
                float v = a[r] + bv;
                out[(size_t)(mbase + r) * OUT_N + col] = v > 0.f ? v : 0.f;
            }
        }
    }
}

// ---------------------------------------------------------------------------
// Inputs (setup_inputs order): x0, x1, s1, s2, W, b, h1, h2
// Workspace: z bf16 [512,16000] = 16,384,000 bytes.
// ---------------------------------------------------------------------------
extern "C" void kernel_launch(void* const* d_in, const int* in_sizes, int n_in,
                              void* d_out, int out_size, void* d_ws, size_t ws_size,
                              hipStream_t stream)
{
    const float* x0   = (const float*)d_in[0];
    const float* x1   = (const float*)d_in[1];
    const float* s1   = (const float*)d_in[2];
    const float* s2   = (const float*)d_in[3];
    const float* W    = (const float*)d_in[4];
    const float* bias = (const float*)d_in[5];
    const int*   h1   = (const int*)d_in[6];
    const int*   h2   = (const int*)d_in[7];
    float*  out = (float*)d_out;
    __bf16* z   = (__bf16*)d_ws;

    dim3 g1(4, B_SZ);
    mcb_conv_kernel<<<g1, dim3(256), 0, stream>>>(x0, x1, s1, s2, h1, h2, z);

    dim3 g2((OUT_N + 127) / 128, B_SZ / 128);
    mcb_gemm_kernel<<<g2, dim3(256), 0, stream>>>(z, W, bias, out);
}
